// InterContrastiveLossDNS_23356032155979
// MI455X (gfx1250) — matrix operand order repaired and checked by reading, not compiled, not run on hardware
//
#include <hip/hip_runtime.h>
#include <hip/hip_bf16.h>
#include <math.h>

// Static problem sizes
#define B_   32
#define C_   512
#define N_   64
#define NN_  4096
#define S_   128
#define M_   256
#define P_   2080
#define PT_  130          // P_/16 proposal tiles (exact)
#define KC_  16           // C_/32 k-chunks
#define TINV 10.0f        // 1/T_TEMP

typedef __attribute__((ext_vector_type(16))) __bf16 v16bf;
typedef __attribute__((ext_vector_type(8)))  float  v8f;
typedef __attribute__((ext_vector_type(4)))  unsigned int u32x4;

union BFrag { v16bf v; u32x4 q[2]; unsigned int u[8]; unsigned short s[16]; };

__device__ __forceinline__ unsigned short f2bf(float f) {
  unsigned int x = __float_as_uint(f);
  x += 0x7FFFu + ((x >> 16) & 1u);           // round-to-nearest-even
  return (unsigned short)(x >> 16);
}

__device__ __forceinline__ float rsum16(float v) {   // reduce within 16-lane half
  v += __shfl_xor(v, 1, 32);
  v += __shfl_xor(v, 2, 32);
  v += __shfl_xor(v, 4, 32);
  v += __shfl_xor(v, 8, 32);
  return v;
}

// ---------------- workspace layout (bytes); total < 512 KB ----------------
#define WS_FIDX   0u         // int[P_]
#define WS_RSNORM 16384u     // float[S_]
#define WS_SB     20480u     // ushort bf16[S_*C_]  (128 KB), 16B aligned
#define WS_VNORM  163840u    // float[B_*P_]        (260 KB)
#define WS_TOPP   458752u    // int[M_]
#define WS_TOT    462848u    // float[S_]
#define WS_POS    463872u    // float[S_]
#define WS_ACC    464896u    // float[2]

// ---------------------------------------------------------------------------
__global__ void k_init(int* __restrict__ fidx, float* __restrict__ tot,
                       float* __restrict__ pos, float* __restrict__ accL) {
  int t = threadIdx.x;
  if (t < S_) { tot[t] = 0.f; pos[t] = 0.f; }
  if (t < 2) accL[t] = 0.f;
  for (int p = t; p < P_; p += 256) {
    int i = 0, start = 0;
    for (int r = 0; r < N_; ++r) {
      int next = start + (N_ - r);
      if (p < next) { i = r; break; }
      start = next;
    }
    int j = i + (p - start);
    fidx[p] = i * N_ + j;                    // flat upper-tri index, row-major
  }
}

// sentence norms (reciprocal) + bf16 conversion
__global__ void k_sents(const float* __restrict__ sents,
                        float* __restrict__ rsnorm,
                        unsigned short* __restrict__ SB) {
  int s = blockIdx.x, t = threadIdx.x;
  __shared__ float red[128];
  float acc = 0.f;
  for (int c = t; c < C_; c += 128) {
    float v = sents[s * C_ + c];
    SB[s * C_ + c] = f2bf(v);
    acc += v * v;
  }
  red[t] = acc; __syncthreads();
  for (int o = 64; o > 0; o >>= 1) {
    if (t < o) red[t] += red[t + o];
    __syncthreads();
  }
  if (t == 0) rsnorm[s] = 1.f / fmaxf(sqrtf(red[0]), 1e-12f);
}

// top-1 proposal per moment over the gathered IoUs (first-max tie break)
__global__ void k_topk(const float* __restrict__ iou2ds,
                       const int* __restrict__ fidx, int* __restrict__ topp) {
  int m = blockIdx.x, t = threadIdx.x;
  __shared__ float bv[128]; __shared__ int bi[128];
  float best = -1.f; int bidx = 0;
  for (int p = t; p < P_; p += 128) {
    float u = iou2ds[(size_t)m * NN_ + fidx[p]];
    if (u > best) { best = u; bidx = p; }
  }
  bv[t] = best; bi[t] = bidx; __syncthreads();
  for (int o = 64; o > 0; o >>= 1) {
    if (t < o) {
      if (bv[t + o] > bv[t] || (bv[t + o] == bv[t] && bi[t + o] < bi[t])) {
        bv[t] = bv[t + o]; bi[t] = bi[t + o];
      }
    }
    __syncthreads();
  }
  if (t == 0) topp[m] = bi[0];
}

// main streamed GEMM: sents(128xC) x v(16-proposal tile)^T, fused norms + LSE
// A fragments: direct b128 loads from L2-resident SB (no LDS staging).
// B fragments: double-buffered LDS tile, two ds_load_b128 per lane.
// Video gather: register-staged + software-pipelined (loads for chunk k+1 in
// flight during chunk k's WMMA), non-temporal (single-use 268 MB stream).
__global__ __launch_bounds__(256) void k_query(
    const float* __restrict__ video, const float* __restrict__ iou2d,
    const unsigned short* __restrict__ SB, const float* __restrict__ rsnorm,
    const int* __restrict__ fidx, float* __restrict__ vnorm,
    float* __restrict__ tot, float* __restrict__ pos) {
  const int b   = blockIdx.x / PT_;
  const int p0  = (blockIdx.x % PT_) * 16;
  const int tid = threadIdx.x, lane = tid & 31, w = tid >> 5;
  const int n = lane & 15, hi = lane >> 4;

  __shared__ unsigned short vb[2][16][40];   // [buf][p][k] bf16; 80B rows (16B aligned, stride 20 dwords)
  __shared__ float vn[16];
  __shared__ int   fx[16];

  if (tid < 16) { fx[tid] = fidx[p0 + tid]; vn[tid] = 0.f; }
  __syncthreads();

  const int pl = tid & 15;                   // proposal lane for staging
  const int c0 = tid >> 4;                   // 0..15 (k within chunk half)
  const size_t vbase = (size_t)b * (C_ * NN_) + (size_t)fx[pl];
  const u32x4* SB4 = (const u32x4*)SB;
  const int srow = w * 16 + n;               // this lane's sentence row

  float f0 = __builtin_nontemporal_load(&video[vbase + (size_t)c0 * NN_]);
  float f1 = __builtin_nontemporal_load(&video[vbase + (size_t)(c0 + 16) * NN_]);

  v8f acc = {0.f, 0.f, 0.f, 0.f, 0.f, 0.f, 0.f, 0.f};
  float vacc = 0.f;                          // thread-private ||v||^2 partial

  for (int kc = 0; kc < KC_; ++kc) {
    const int buf = kc & 1;
    vacc += f0 * f0 + f1 * f1;
    vb[buf][pl][c0]      = f2bf(f0);
    vb[buf][pl][c0 + 16] = f2bf(f1);
    __syncthreads();                         // stores to buf visible

    if (kc + 1 < KC_) {                      // prefetch next chunk during math
      const int K0n = (kc + 1) * 32;
      f0 = __builtin_nontemporal_load(&video[vbase + (size_t)(K0n + c0) * NN_]);
      f1 = __builtin_nontemporal_load(&video[vbase + (size_t)(K0n + c0 + 16) * NN_]);
    }

    BFrag A, Bf;
    // A (16x32 bf16, rows = sentences): two aligned b128 from global SB
    A.q[0] = SB4[srow * 64 + kc * 4 + hi];
    A.q[1] = SB4[srow * 64 + kc * 4 + 2 + hi];
    // B (32x16 bf16, cols = proposals): two aligned b128 from LDS, row = p
    const u32x4* vr = (const u32x4*)(&vb[buf][n][0]);
    Bf.q[0] = vr[hi];
    Bf.q[1] = vr[2 + hi];
    acc = __builtin_amdgcn_wmma_f32_16x16x32_bf16(
        false, A.v, false, Bf.v, (short)0, acc, false, false);
    // next iteration stores to the other buffer; the barrier above makes the
    // one-barrier-per-chunk double-buffer scheme race-free
  }

  atomicAdd(&vn[pl], vacc);
  __syncthreads();

  const float rv = 1.f / fmaxf(sqrtf(vn[n]), 1e-12f);
  if (tid < 16) vnorm[(size_t)b * P_ + p0 + tid] = vn[tid];

  #pragma unroll
  for (int r = 0; r < 8; ++r) {
    int s = w * 16 + r + 8 * hi;             // D row -> sentence
    float scaled = acc[r] * rv * rsnorm[s];  // cosine similarity
    float e = __expf(scaled * TINV);
    float red = rsum16(e);
    if (n == 0) atomicAdd(&tot[s], red);
    // positive-mask sum: own-video block with IoU > 0.5
    bool mine = (b == (s >> 2));
    float pe = 0.f;
    if (mine) {
      float iou = iou2d[(size_t)s * NN_ + fx[n]];
      if (iou > 0.5f) pe = e;
    }
    pe = rsum16(pe);
    if (mine && n == 0) atomicAdd(&pos[s], pe);
  }
}

// inter-video GEMM (top-1 proposals x sentences) + both loss epilogues
__global__ __launch_bounds__(256) void k_video(
    const float* __restrict__ video,
    const unsigned short* __restrict__ SB, const float* __restrict__ rsnorm,
    const int* __restrict__ fidx, const int* __restrict__ topp,
    const float* __restrict__ vnorm,
    const float* __restrict__ tot, const float* __restrict__ pos,
    float* __restrict__ accL) {
  const int mt  = blockIdx.x;                // 16-moment tile
  const int tid = threadIdx.x, lane = tid & 31, w = tid >> 5;
  const int n = lane & 15, hi = lane >> 4;

  __shared__ unsigned short tv[2][16][40];   // [buf][m][k] bf16, 80B rows
  __shared__ int   tf[16], bmv[16];
  __shared__ float rvm[16], ivsum[16], posv[16];

  if (tid < 16) {
    int m  = mt * 16 + tid;
    int tp = topp[m];
    tf[tid]  = fidx[tp];
    int bm   = m >> 3;                       // scatter_m2v
    bmv[tid] = bm;
    rvm[tid] = 1.f / fmaxf(sqrtf(vnorm[(size_t)bm * P_ + tp]), 1e-12f);
    ivsum[tid] = 0.f;
  }
  __syncthreads();

  const int ml = tid & 15, k0r = tid >> 4;
  const size_t tbase = (size_t)bmv[ml] * (C_ * NN_) + (size_t)tf[ml];
  const u32x4* SB4 = (const u32x4*)SB;
  const int srow = w * 16 + n;

  float f0 = video[tbase + (size_t)k0r * NN_];
  float f1 = video[tbase + (size_t)(k0r + 16) * NN_];

  v8f acc = {0.f, 0.f, 0.f, 0.f, 0.f, 0.f, 0.f, 0.f};

  for (int kc = 0; kc < KC_; ++kc) {
    const int buf = kc & 1;
    tv[buf][ml][k0r]      = f2bf(f0);
    tv[buf][ml][k0r + 16] = f2bf(f1);
    __syncthreads();

    if (kc + 1 < KC_) {
      const int K0n = (kc + 1) * 32;
      f0 = video[tbase + (size_t)(K0n + k0r) * NN_];
      f1 = video[tbase + (size_t)(K0n + k0r + 16) * NN_];
    }

    BFrag A, Bf;
    // A rows = moments: two aligned b128 from LDS
    const u32x4* tr = (const u32x4*)(&tv[buf][n][0]);
    A.q[0] = tr[hi];
    A.q[1] = tr[2 + hi];
    // B cols = this wave's 16-sentence stripe: two aligned b128 from L2
    Bf.q[0] = SB4[srow * 64 + kc * 4 + hi];
    Bf.q[1] = SB4[srow * 64 + kc * 4 + 2 + hi];
    acc = __builtin_amdgcn_wmma_f32_16x16x32_bf16(
        false, A.v, false, Bf.v, (short)0, acc, false, false);
  }

  #pragma unroll
  for (int r = 0; r < 8; ++r) {
    int mloc = r + 8 * hi;                   // local moment
    int s    = w * 16 + n;                   // sentence (D column)
    float scaled = acc[r] * rvm[mloc] * rsnorm[s];
    float e = __expf(scaled * TINV);
    float red = rsum16(e);                   // sum over this wave's 16 sentences
    if (n == 0) atomicAdd(&ivsum[mloc], red);
    int m = mt * 16 + mloc;
    if (s == (m >> 1)) posv[mloc] = scaled;  // exactly one lane matches
  }
  __syncthreads();

  if (tid < 16) {
    int m = mt * 16 + tid;
    float p10 = posv[tid] * TINV;
    float pe  = __expf(p10);
    float negIV = ivsum[tid] - pe;           // exclude s == m/2
    float lIV = -(p10 - logf(pe + negIV));
    int sm = m >> 1;
    float negQ = tot[sm] - pos[sm];          // total minus pos-masked sum
    float lIQ = -(p10 - logf(pe + negQ));
    atomicAdd(&accL[0], lIV);
    atomicAdd(&accL[1], lIQ);
  }
}

__global__ void k_final(const float* __restrict__ accL, float* __restrict__ out) {
  if (threadIdx.x == 0 && blockIdx.x == 0) {
    float liv = accL[0] / (float)M_;
    float liq = accL[1] / (float)M_;
    out[0] = liv + liq;
    out[1] = liv;
    out[2] = liq;
  }
}

extern "C" void kernel_launch(void* const* d_in, const int* in_sizes, int n_in,
                              void* d_out, int out_size, void* d_ws, size_t ws_size,
                              hipStream_t stream) {
  (void)in_sizes; (void)n_in; (void)out_size; (void)ws_size;
  const float* video  = (const float*)d_in[0];   // (B,C,N,N)
  const float* sents  = (const float*)d_in[1];   // (S,C)
  // d_in[2]=num_sentences (const 4), d_in[3]=num_targets (const 2): folded statically
  const float* iou2d  = (const float*)d_in[4];   // (S,N,N)
  const float* iou2ds = (const float*)d_in[5];   // (M,N,N)
  // d_in[6]=mask2d (static upper-tri), d_in[7]=epoch (DNS branch inactive)

  char* ws = (char*)d_ws;   // needs ~455 KB of scratch
  int*            fidx   = (int*)(ws + WS_FIDX);
  float*          rsnorm = (float*)(ws + WS_RSNORM);
  unsigned short* SB     = (unsigned short*)(ws + WS_SB);
  float*          vnorm  = (float*)(ws + WS_VNORM);
  int*            topp   = (int*)(ws + WS_TOPP);
  float*          tot    = (float*)(ws + WS_TOT);
  float*          pos    = (float*)(ws + WS_POS);
  float*          accL   = (float*)(ws + WS_ACC);

  k_init <<<1,   256, 0, stream>>>(fidx, tot, pos, accL);
  k_sents<<<S_,  128, 0, stream>>>(sents, rsnorm, SB);
  k_topk <<<M_,  128, 0, stream>>>(iou2ds, fidx, topp);
  k_query<<<B_ * PT_, 256, 0, stream>>>(video, iou2d, SB, rsnorm, fidx,
                                        vnorm, tot, pos);
  k_video<<<M_ / 16, 256, 0, stream>>>(video, SB, rsnorm, fidx, topp,
                                       vnorm, tot, pos, accL);
  k_final<<<1, 64, 0, stream>>>(accL, (float*)d_out);
}
